// GeomMultiTokenTransformer_84413287236171
// MI455X (gfx1250) — compile-verified, matrix operand order checked by reading, not run on hardware
//
#include <hip/hip_runtime.h>

typedef __bf16 bf16_t;
typedef __attribute__((ext_vector_type(16))) __bf16 v16bf;
typedef __attribute__((ext_vector_type(8)))  __bf16 v8bf;
typedef __attribute__((ext_vector_type(8)))  float   v8f;

union AFrag { v16bf v; v8bf h[2]; };

#define WMMA_BF16(a, b, c) \
  __builtin_amdgcn_wmma_f32_16x16x32_bf16(false, (a), false, (b), (short)0, (c), false, false)

// CDNA5 async copy global -> LDS (ASYNCcnt-tracked, no VGPR staging).
__device__ __forceinline__ void async_load_b128(unsigned lds_off, const void* gaddr) {
  asm volatile("global_load_async_to_lds_b128 %0, %1, off"
               :: "v"(lds_off), "v"(gaddr) : "memory");
}

__device__ __forceinline__ float red16_max(float v) {
  v = fmaxf(v, __shfl_xor(v, 1, 16));
  v = fmaxf(v, __shfl_xor(v, 2, 16));
  v = fmaxf(v, __shfl_xor(v, 4, 16));
  v = fmaxf(v, __shfl_xor(v, 8, 16));
  return v;
}
__device__ __forceinline__ float red16_sum(float v) {
  v += __shfl_xor(v, 1, 16);
  v += __shfl_xor(v, 2, 16);
  v += __shfl_xor(v, 4, 16);
  v += __shfl_xor(v, 8, 16);
  return v;
}

// ---------------------------------------------------------------------------
// f32 -> bf16 converts (straight + transposing, for weight B^T layout)
// ---------------------------------------------------------------------------
__global__ void convert_kernel(const float* __restrict__ src, bf16_t* __restrict__ dst, int n) {
  int i = blockIdx.x * 256 + threadIdx.x;
  if (i < n) dst[i] = (bf16_t)src[i];
}

__global__ void convertT_kernel(const float* __restrict__ src, bf16_t* __restrict__ dst,
                                int K, int Nn) {
  int i = blockIdx.x * 256 + threadIdx.x;
  if (i < K * Nn) {
    int k = i / Nn, n = i - k * Nn;
    dst[(size_t)n * K + k] = (bf16_t)src[i];
  }
}

// ---------------------------------------------------------------------------
// Tiled bf16 WMMA GEMM:  C[M x Nn] = A[M x K] * B[K x Nn] + bias
//   A row-major bf16, Bt = B^T row-major bf16 ([Nn x K]).
//   Block tile 128x128, 8 waves, each wave 32x64 (2x4 WMMA tiles), K-step 32.
//   Double-buffered LDS filled by CDNA5 async global->LDS copies; branch-free
//   steady state with unconditional s_wait_asynccnt 0x4.
// ---------------------------------------------------------------------------
__global__ __launch_bounds__(256, 2) void gemm_bf16_kernel(
    const bf16_t* __restrict__ A, const bf16_t* __restrict__ Bt,
    const float* __restrict__ bias,
    float* __restrict__ outF, bf16_t* __restrict__ outB, bf16_t* __restrict__ outVt,
    int M, int Nn, int K, int relu) {
  __shared__ __align__(16) bf16_t As[2][128][40];
  __shared__ __align__(16) bf16_t Bs[2][128][40];
  const int tid = threadIdx.x;
  const int wave = tid >> 5, lane = tid & 31;
  const int lane16 = lane & 15, hs = lane >> 4;
  const int m0 = blockIdx.y * 128, n0 = blockIdx.x * 128;
  const int mw = (wave & 3) * 32, nw = (wave >> 2) * 64;

  // This thread's two 16-byte chunks of the 128x32 tile (512 chunks / 256 thr).
  const int id0 = tid * 2, id1 = tid * 2 + 1;
  const int row0 = id0 >> 2, ch0 = (id0 & 3) * 8;
  const int row1 = id1 >> 2, ch1 = (id1 & 3) * 8;

  // Per-thread base pointers / LDS offsets, fixed across k-steps.
  const bf16_t* gA0 = &A[(size_t)(m0 + row0) * K + ch0];
  const bf16_t* gA1 = &A[(size_t)(m0 + row1) * K + ch1];
  const bf16_t* gB0 = &Bt[(size_t)(n0 + row0) * K + ch0];
  const bf16_t* gB1 = &Bt[(size_t)(n0 + row1) * K + ch1];
  const unsigned lA0[2] = {(unsigned)(size_t)&As[0][row0][ch0], (unsigned)(size_t)&As[1][row0][ch0]};
  const unsigned lA1[2] = {(unsigned)(size_t)&As[0][row1][ch1], (unsigned)(size_t)&As[1][row1][ch1]};
  const unsigned lB0[2] = {(unsigned)(size_t)&Bs[0][row0][ch0], (unsigned)(size_t)&Bs[1][row0][ch0]};
  const unsigned lB1[2] = {(unsigned)(size_t)&Bs[0][row1][ch1], (unsigned)(size_t)&Bs[1][row1][ch1]};

  auto issue_tile = [&](int ks, int buf) {
    const int k0 = ks << 5;
    async_load_b128(lA0[buf], gA0 + k0);
    async_load_b128(lA1[buf], gA1 + k0);
    async_load_b128(lB0[buf], gB0 + k0);
    async_load_b128(lB1[buf], gB1 + k0);
  };

  v8f acc[2][4] = {};
  auto compute = [&](int buf) {
    AFrag af[2], bfrg[4];
#pragma unroll
    for (int i = 0; i < 2; ++i) {
      af[i].h[0] = *(const v8bf*)&As[buf][mw + 16 * i + lane16][hs * 8];
      af[i].h[1] = *(const v8bf*)&As[buf][mw + 16 * i + lane16][16 + hs * 8];
    }
#pragma unroll
    for (int j = 0; j < 4; ++j) {
      bfrg[j].h[0] = *(const v8bf*)&Bs[buf][nw + 16 * j + lane16][hs * 8];
      bfrg[j].h[1] = *(const v8bf*)&Bs[buf][nw + 16 * j + lane16][16 + hs * 8];
    }
#pragma unroll
    for (int i = 0; i < 2; ++i)
#pragma unroll
      for (int j = 0; j < 4; ++j)
        acc[i][j] = WMMA_BF16(af[i].v, bfrg[j].v, acc[i][j]);
  };

  const int nk = K >> 5;
  issue_tile(0, 0);
  for (int ks = 0; ks < nk - 1; ++ks) {        // branch-free steady state
    const int buf = ks & 1;
    issue_tile(ks + 1, buf ^ 1);
    asm volatile("s_wait_asynccnt 0x4" ::: "memory");  // tile ks landed (ours)
    __syncthreads();                            // everyone's writes visible
    compute(buf);
    __syncthreads();                            // reads done before refill
  }
  asm volatile("s_wait_asynccnt 0x0" ::: "memory");     // last tile landed
  __syncthreads();
  compute((nk - 1) & 1);

#pragma unroll
  for (int i = 0; i < 2; ++i) {
#pragma unroll
    for (int j = 0; j < 4; ++j) {
      const int n_g = n0 + nw + 16 * j + lane16;
      const float bv = bias ? bias[n_g] : 0.f;
#pragma unroll
      for (int r = 0; r < 8; ++r) {
        const int m_g = m0 + mw + 16 * i + r + 8 * hs;
        float v = acc[i][j][r] + bv;
        if (relu) v = fmaxf(v, 0.f);
        if (outF) outF[(size_t)m_g * Nn + n_g] = v;
        if (outB) outB[(size_t)m_g * Nn + n_g] = (bf16_t)v;
        if (outVt) {  // write V transposed: [B,H,DK,N] with N=1024, D=512
          int bb = m_g >> 10, nn = m_g & 1023;
          int hh = n_g >> 6, dk = n_g & 63;
          outVt[((size_t)((bb * 8 + hh) * 64 + dk) << 10) + nn] = (bf16_t)v;
        }
      }
    }
  }
}

// ---------------------------------------------------------------------------
// Flash-style attention with geometric bias alpha * <coords_q, coords_k>.
// grid = B*H*(N/64), block = 128 (4 waves); wave owns a 16-query tile.
// ---------------------------------------------------------------------------
__global__ __launch_bounds__(128) void attn_kernel(
    const bf16_t* __restrict__ Q, const bf16_t* __restrict__ Km,
    const bf16_t* __restrict__ Vt, const float* __restrict__ coords,
    const float* __restrict__ alpha_p, int layer, float* __restrict__ outO) {
  __shared__ __align__(16) bf16_t pb[4][16][40];
  const int wave = threadIdx.x >> 5, lane = threadIdx.x & 31;
  const int lane16 = lane & 15, hs = lane >> 4;
  const int qt = blockIdx.x & 15, bh = blockIdx.x >> 4;
  const int b = bh >> 3, h = bh & 7;
  const int q0 = qt * 64 + wave * 16;
  const float alpha = alpha_p[layer];
  const float scale = 0.125f;  // 1/sqrt(64)

  AFrag qa[2];
  {
    const bf16_t* qrow = Q + ((size_t)(b * 1024 + q0 + lane16) * 512) + h * 64;
#pragma unroll
    for (int t = 0; t < 2; ++t) {
      qa[t].h[0] = *(const v8bf*)(qrow + t * 32 + hs * 8);
      qa[t].h[1] = *(const v8bf*)(qrow + t * 32 + 16 + hs * 8);
    }
  }
  float cq[8][3];
#pragma unroll
  for (int r = 0; r < 8; ++r) {
    const float* cp = coords + (size_t)(b * 1024 + q0 + r + 8 * hs) * 3;
    cq[r][0] = cp[0]; cq[r][1] = cp[1]; cq[r][2] = cp[2];
  }
  float mrun[8], lrun[8];
#pragma unroll
  for (int r = 0; r < 8; ++r) { mrun[r] = -3.0e38f; lrun[r] = 0.f; }
  v8f acc[4] = {};

  for (int kb = 0; kb < 32; ++kb) {
    const int k0 = kb * 32;
    v8f s[2] = {};
#pragma unroll
    for (int j = 0; j < 2; ++j) {
      const bf16_t* krow = Km + ((size_t)(b * 1024 + k0 + j * 16 + lane16) * 512) + h * 64;
#pragma unroll
      for (int t = 0; t < 2; ++t) {
        AFrag kf;
        kf.h[0] = *(const v8bf*)(krow + t * 32 + hs * 8);
        kf.h[1] = *(const v8bf*)(krow + t * 32 + 16 + hs * 8);
        s[j] = WMMA_BF16(qa[t].v, kf.v, s[j]);
      }
    }
    float ck[2][3];
#pragma unroll
    for (int j = 0; j < 2; ++j) {
      const float* cp = coords + (size_t)(b * 1024 + k0 + j * 16 + lane16) * 3;
      ck[j][0] = cp[0]; ck[j][1] = cp[1]; ck[j][2] = cp[2];
    }
    float cfac[8];
#pragma unroll
    for (int r = 0; r < 8; ++r) {
      float g0 = cq[r][0] * ck[0][0] + cq[r][1] * ck[0][1] + cq[r][2] * ck[0][2];
      float g1 = cq[r][0] * ck[1][0] + cq[r][1] * ck[1][1] + cq[r][2] * ck[1][2];
      float s0 = s[0][r] * scale + alpha * g0;
      float s1 = s[1][r] * scale + alpha * g1;
      float bm = red16_max(fmaxf(s0, s1));
      float mnew = fmaxf(mrun[r], bm);
      float c = __expf(mrun[r] - mnew);
      float p0 = __expf(s0 - mnew);
      float p1 = __expf(s1 - mnew);
      lrun[r] = lrun[r] * c + red16_sum(p0 + p1);
      mrun[r] = mnew;
      cfac[r] = c;
      const int m = r + 8 * hs;
      pb[wave][m][lane16]      = (bf16_t)p0;   // transpose P through per-wave LDS
      pb[wave][m][16 + lane16] = (bf16_t)p1;
    }
#pragma unroll
    for (int j = 0; j < 4; ++j)
#pragma unroll
      for (int r = 0; r < 8; ++r) acc[j][r] *= cfac[r];

    asm volatile("s_wait_dscnt 0" ::: "memory");  // per-wave LDS ordering
    AFrag pa;
    pa.h[0] = *(const v8bf*)&pb[wave][lane16][hs * 8];
    pa.h[1] = *(const v8bf*)&pb[wave][lane16][16 + hs * 8];
#pragma unroll
    for (int j = 0; j < 4; ++j) {
      const bf16_t* vrow = Vt + ((size_t)((b * 8 + h) * 64 + j * 16 + lane16) << 10) + k0;
      AFrag vf;
      vf.h[0] = *(const v8bf*)(vrow + hs * 8);
      vf.h[1] = *(const v8bf*)(vrow + 16 + hs * 8);
      acc[j] = WMMA_BF16(pa.v, vf.v, acc[j]);
    }
  }
#pragma unroll
  for (int r = 0; r < 8; ++r) {
    const float inv = 1.f / lrun[r];
    const int row = b * 1024 + q0 + r + 8 * hs;
#pragma unroll
    for (int j = 0; j < 4; ++j)
      outO[(size_t)row * 512 + h * 64 + j * 16 + lane16] = acc[j][r] * inv;
  }
}

// ---------------------------------------------------------------------------
// Fused residual + LayerNorm; one wave per row of D=512. Writes f32 and bf16.
// ---------------------------------------------------------------------------
__global__ __launch_bounds__(256) void ln_kernel(
    const float* __restrict__ xin, const float* __restrict__ add,
    const float* __restrict__ g, const float* __restrict__ be,
    float* __restrict__ outF, bf16_t* __restrict__ outB, int rows) {
  const int wave = threadIdx.x >> 5, lane = threadIdx.x & 31;
  const int row = blockIdx.x * 8 + wave;
  if (row >= rows) return;
  const float* xr = xin + (size_t)row * 512;
  const float* ar = add + (size_t)row * 512;
  float v[16], s = 0.f, s2 = 0.f;
#pragma unroll
  for (int k = 0; k < 16; ++k) {
    int c = lane + 32 * k;
    float t = xr[c] + ar[c];
    v[k] = t; s += t; s2 += t * t;
  }
#pragma unroll
  for (int o = 16; o >= 1; o >>= 1) {
    s += __shfl_xor(s, o, 32);
    s2 += __shfl_xor(s2, o, 32);
  }
  float mean = s * (1.f / 512.f);
  float var = s2 * (1.f / 512.f) - mean * mean;
  float rs = rsqrtf(var + 1e-5f);
#pragma unroll
  for (int k = 0; k < 16; ++k) {
    int c = lane + 32 * k;
    float y = (v[k] - mean) * rs * g[c] + be[c];
    if (outF) outF[(size_t)row * 512 + c] = y;
    if (outB) outB[(size_t)row * 512 + c] = (bf16_t)y;
  }
}

// ---------------------------------------------------------------------------
extern "C" void kernel_launch(void* const* d_in, const int* in_sizes, int n_in,
                              void* d_out, int out_size, void* d_ws, size_t ws_size,
                              hipStream_t stream) {
  const float* x      = (const float*)d_in[0];
  const float* coords = (const float*)d_in[1];
  const float* Wq = (const float*)d_in[2];
  const float* bq = (const float*)d_in[3];
  const float* Wk = (const float*)d_in[4];
  const float* bk = (const float*)d_in[5];
  const float* Wv = (const float*)d_in[6];
  const float* bv = (const float*)d_in[7];
  const float* alpha = (const float*)d_in[8];
  const float* W1 = (const float*)d_in[9];
  const float* b1 = (const float*)d_in[10];
  const float* W2 = (const float*)d_in[11];
  const float* b2 = (const float*)d_in[12];
  const float* g1  = (const float*)d_in[13];
  const float* be1 = (const float*)d_in[14];
  const float* g2  = (const float*)d_in[15];
  const float* be2 = (const float*)d_in[16];
  (void)in_sizes; (void)n_in; (void)out_size; (void)ws_size;

  constexpr int Bc = 4, Nn = 1024, D = 512, H = 8, DF = 2048, Lc = 4;
  constexpr int M = Bc * Nn;  // 4096

  char* ws = (char*)d_ws;
  size_t off = 0;
  auto alloc = [&](size_t bytes) -> void* {
    void* p = ws + off;
    off = (off + bytes + 255) & ~(size_t)255;
    return p;
  };
  bf16_t* WqT = (bf16_t*)alloc((size_t)D * D * 2);
  bf16_t* WkT = (bf16_t*)alloc((size_t)D * D * 2);
  bf16_t* WvT = (bf16_t*)alloc((size_t)D * D * 2);
  bf16_t* W1T = (bf16_t*)alloc((size_t)D * DF * 2);
  bf16_t* W2T = (bf16_t*)alloc((size_t)DF * D * 2);
  bf16_t* Xb  = (bf16_t*)alloc((size_t)M * D * 2);
  bf16_t* Qb  = (bf16_t*)alloc((size_t)M * D * 2);
  bf16_t* Kb  = (bf16_t*)alloc((size_t)M * D * 2);
  bf16_t* Vtb = (bf16_t*)alloc((size_t)M * D * 2);
  bf16_t* x1b = (bf16_t*)alloc((size_t)M * D * 2);
  bf16_t* hb  = (bf16_t*)alloc((size_t)M * DF * 2);
  float* attn = (float*)alloc((size_t)M * D * 4);
  float* ffo  = (float*)alloc((size_t)M * D * 4);
  float* x1f  = (float*)alloc((size_t)M * D * 4);
  float* xcur = (float*)alloc((size_t)M * D * 4);

  convert_kernel<<<(M * D + 255) / 256, 256, 0, stream>>>(x, Xb, M * D);

  for (int l = 0; l < Lc; ++l) {
    convertT_kernel<<<(D * D + 255) / 256, 256, 0, stream>>>(Wq + (size_t)l * D * D, WqT, D, D);
    convertT_kernel<<<(D * D + 255) / 256, 256, 0, stream>>>(Wk + (size_t)l * D * D, WkT, D, D);
    convertT_kernel<<<(D * D + 255) / 256, 256, 0, stream>>>(Wv + (size_t)l * D * D, WvT, D, D);
    convertT_kernel<<<(D * DF + 255) / 256, 256, 0, stream>>>(W1 + (size_t)l * D * DF, W1T, D, DF);
    convertT_kernel<<<(DF * D + 255) / 256, 256, 0, stream>>>(W2 + (size_t)l * DF * D, W2T, DF, D);

    dim3 g512(D / 128, M / 128);
    gemm_bf16_kernel<<<g512, 256, 0, stream>>>(Xb, WqT, bq + l * D, nullptr, Qb, nullptr, M, D, D, 0);
    gemm_bf16_kernel<<<g512, 256, 0, stream>>>(Xb, WkT, bk + l * D, nullptr, Kb, nullptr, M, D, D, 0);
    gemm_bf16_kernel<<<g512, 256, 0, stream>>>(Xb, WvT, bv + l * D, nullptr, nullptr, Vtb, M, D, D, 0);

    attn_kernel<<<Bc * H * (Nn / 64), 128, 0, stream>>>(Qb, Kb, Vtb, coords, alpha, l, attn);

    const float* xin = (l == 0) ? x : xcur;
    ln_kernel<<<M / 8, 256, 0, stream>>>(xin, attn, g1 + l * D, be1 + l * D, x1f, x1b, M);

    dim3 gff1(DF / 128, M / 128);
    gemm_bf16_kernel<<<gff1, 256, 0, stream>>>(x1b, W1T, b1 + l * DF, nullptr, hb, nullptr, M, DF, D, 1);
    gemm_bf16_kernel<<<g512, 256, 0, stream>>>(hb, W2T, b2 + l * D, ffo, nullptr, nullptr, M, D, DF, 0);

    float* xoutF = (l == Lc - 1) ? (float*)d_out : xcur;
    bf16_t* xoutB = (l == Lc - 1) ? nullptr : Xb;
    ln_kernel<<<M / 8, 256, 0, stream>>>(x1f, ffo, g2 + l * D, be2 + l * D, xoutF, xoutB, M);
  }
}
